// CSDSA_53833120088361
// MI455X (gfx1250) — matrix operand order repaired and checked by reading, not compiled, not run on hardware
//
#include <hip/hip_runtime.h>

// ---------------------------------------------------------------------------
// Criss-cross attention for MI455X (gfx1250, wave32).
// All matmuls via V_WMMA_F32_16X16X32_BF16 (bf16 operands, f32 accumulate).
// One wave computes one 16x16 tile; blockDim=256 = 8 waves.
// Data layouts chosen so every WMMA fragment has contiguous K (vector loads);
// weights pre-converted to bf16 so the hot GEMM loop is pure load->wmma.
// ---------------------------------------------------------------------------

typedef __attribute__((ext_vector_type(16))) __bf16 v16bf;
typedef __attribute__((ext_vector_type(8)))  __bf16 v8bf;
typedef __attribute__((ext_vector_type(8)))  float  v8f;
typedef __attribute__((ext_vector_type(16))) float  v16f;

#define B_   8
#define C_   512
#define CQ_  64
#define H_   96
#define W_   96
#define HW_  (H_ * W_)
#define NEGC 1e30f

// --- WMMA wrapper -----------------------------------------------------------
__device__ __forceinline__ v8f wmma_bf16(v16bf a, v16bf b, v8f c) {
  return __builtin_amdgcn_wmma_f32_16x16x32_bf16(false, a, false, b,
                                                 (short)0, c, false, false);
}

// --- Fragment loaders, contiguous-K only (sk == 1) --------------------------
// A (16x32 MxK): lane row m = lane&15, half hi = lane>>4.
// Lane's elements: e=0..15 -> k = hi*16+e (one contiguous run).
__device__ __forceinline__ v16bf ldA_bfc(const __bf16* __restrict__ p, long sm,
                                         int lane) {
  const __bf16* q = p + (long)(lane & 15) * sm + ((lane >> 4) << 4);
  v8bf lo = *(const v8bf*)q;          // 16B
  v8bf hi = *(const v8bf*)(q + 8);    // 16B
  v16bf r;
#pragma unroll
  for (int e = 0; e < 8; ++e) { r[e] = lo[e]; r[8 + e] = hi[e]; }
  return r;
}
// B (32x16 KxN): lane col n = lane&15; k = hi*16 + e (one contiguous run).
__device__ __forceinline__ v16bf ldB_bfc(const __bf16* __restrict__ p, long sn,
                                         int lane) {
  const __bf16* q = p + (long)(lane & 15) * sn + ((lane >> 4) << 4);
  return *(const v16bf*)q;            // 32B
}
__device__ __forceinline__ v16bf ldB_fc(const float* __restrict__ p, long sn,
                                        int lane) {
  const float* q = p + (long)(lane & 15) * sn + ((lane >> 4) << 4);
  v16f v = *(const v16f*)q;           // 64B
  v16bf r;
#pragma unroll
  for (int e = 0; e < 16; ++e) r[e] = (__bf16)v[e];
  return r;
}

// ---------------------------------------------------------------------------
// Kernel 0a: elementwise fp32 -> bf16 convert (for the projection weights).
// ---------------------------------------------------------------------------
__global__ __launch_bounds__(256) void cvt_kernel(const float* __restrict__ s,
                                                  __bf16* __restrict__ d,
                                                  int n) {
  const int i = blockIdx.x * 256 + threadIdx.x;
  if (i < n) d[i] = (__bf16)s[i];
}

// ---------------------------------------------------------------------------
// Kernel 0b: LDS-tiled transpose+convert:  x (b,c,n) f32 -> Xt (b,n,c) bf16
// 32x32 tiles, 256 threads (32x8), coalesced both sides.
// ---------------------------------------------------------------------------
__global__ __launch_bounds__(256) void transpose_kernel(
    const float* __restrict__ x, __bf16* __restrict__ Xt) {
  __shared__ float tile[32][33];
  const int NTn = HW_ / 32;                       // 288
  const int nt = blockIdx.x % NTn;
  const int t  = blockIdx.x / NTn;
  const int ct = t % (C_ / 32);
  const int b  = t / (C_ / 32);
  const int c0 = ct * 32, n0 = nt * 32;
  const int tx = threadIdx.x & 31, ty = threadIdx.x >> 5;

  const float* X = x + (long)b * C_ * HW_;
#pragma unroll
  for (int i = 0; i < 32; i += 8)
    tile[ty + i][tx] = X[(long)(c0 + ty + i) * HW_ + n0 + tx];
  __syncthreads();
  __bf16* O = Xt + (long)b * HW_ * C_;
#pragma unroll
  for (int i = 0; i < 32; i += 8)
    O[(long)(n0 + ty + i) * C_ + c0 + tx] = (__bf16)tile[tx][ty + i];
}

// ---------------------------------------------------------------------------
// Kernel 1: fused QKV projection.  Y = W (MxC) * Xt^T + bias, per batch.
// m-tile mt: [0,4)=Wq, [4,8)=Wk, [8,40)=Wv.
// Q,K stored channel-last twice: (b,w,h,c) and (b,h,w,c).
// V stored natural (b,c,h,w) and transposed (b,w,c,h).
// ---------------------------------------------------------------------------
__global__ __launch_bounds__(256) void proj_kernel(
    const __bf16* __restrict__ Xt,
    const __bf16* __restrict__ Wqb, const float* __restrict__ bq,
    const __bf16* __restrict__ Wkb, const float* __restrict__ bk,
    const __bf16* __restrict__ Wvb, const float* __restrict__ bv,
    __bf16* __restrict__ Qt1, __bf16* __restrict__ Qt2,
    __bf16* __restrict__ Kt1, __bf16* __restrict__ Kt2,
    __bf16* __restrict__ Vn,  __bf16* __restrict__ Vt) {
  const int lane = threadIdx.x & 31;
  const int wid  = blockIdx.x * 8 + (threadIdx.x >> 5);
  const int NT   = HW_ / 16;                      // 576 n-tiles
  const int nt   = wid % NT;
  const int t    = wid / NT;
  const int mt   = t % 40;
  const int b    = t / 40;

  const __bf16* Wm; const float* bias; int m0;
  if (mt < 4)      { Wm = Wqb; bias = bq; m0 = mt * 16; }
  else if (mt < 8) { Wm = Wkb; bias = bk; m0 = (mt - 4) * 16; }
  else             { Wm = Wvb; bias = bv; m0 = (mt - 8) * 16; }

  const __bf16* Xb = Xt + (long)b * HW_ * C_;
  const int     n0 = nt * 16;

  v8f acc = {};
  for (int k0 = 0; k0 < C_; k0 += 32) {
    if (k0 + 32 < C_)
      __builtin_prefetch(Xb + (long)n0 * C_ + k0 + 32, 0, 1);
    v16bf a  = ldA_bfc(Wm + (long)m0 * C_ + k0, (long)C_, lane);   // (m,k)
    v16bf bm = ldB_bfc(Xb + (long)n0 * C_ + k0, (long)C_, lane);   // (k,n): n*C+k
    acc = wmma_bf16(a, bm, acc);
  }

  const int n  = n0 + (lane & 15);                // pixel index
  const int hi = lane >> 4;
  const int mb = m0 + (hi << 3);
  const int h  = n / W_, w = n % W_;
  if (mt < 8) {
    v8bf pk;
#pragma unroll
    for (int r = 0; r < 8; ++r) pk[r] = (__bf16)(acc[r] + bias[mb + r]);
    __bf16* O1 = (mt < 4) ? Qt1 : Kt1;
    __bf16* O2 = (mt < 4) ? Qt2 : Kt2;
    *(v8bf*)(O1 + ((long)(b * W_ + w) * H_ + h) * CQ_ + mb) = pk;  // (b,w,h,c)
    *(v8bf*)(O2 + ((long)b * HW_ + n) * CQ_ + mb) = pk;            // (b,h,w,c)
  } else {
#pragma unroll
    for (int r = 0; r < 8; ++r) {
      const __bf16 v = (__bf16)(acc[r] + bias[mb + r]);
      const int c = mb + r;
      Vn[((long)b * C_ + c) * HW_ + n] = v;                        // (b,c,h,w)
      Vt[((long)(b * W_ + w) * C_ + c) * H_ + h] = v;              // (b,w,c,h)
    }
  }
}

// ---------------------------------------------------------------------------
// Kernel 2a: energy_H[b,h,w,g] = sum_c q[c,h,w] k[c,g,w]   (per (b,w))
// Q from Qt1 (b,w,h,c); K from Kt1 (b,w,g,c): both contiguous in c.
// ---------------------------------------------------------------------------
__global__ __launch_bounds__(256) void energyH_kernel(
    const __bf16* __restrict__ Qt1, const __bf16* __restrict__ Kt1,
    float* __restrict__ eH) {
  const int lane = threadIdx.x & 31;
  const int wid  = blockIdx.x * 8 + (threadIdx.x >> 5);
  const int tile = wid % 36;
  const int w    = (wid / 36) % W_;
  const int b    = wid / (36 * W_);
  const int h0   = (tile / 6) * 16;
  const int g0   = (tile % 6) * 16;

  const __bf16* Qp = Qt1 + ((long)(b * W_ + w) * H_ + h0) * CQ_;   // A(m=h,k=c)
  const __bf16* Kp = Kt1 + ((long)(b * W_ + w) * H_ + g0) * CQ_;   // B(k=c,n=g)

  v8f acc = {};
#pragma unroll
  for (int k0 = 0; k0 < CQ_; k0 += 32) {
    v16bf a  = ldA_bfc(Qp + k0, (long)CQ_, lane);
    v16bf bm = ldB_bfc(Kp + k0, (long)CQ_, lane);
    acc = wmma_bf16(a, bm, acc);
  }

  const int g  = g0 + (lane & 15);
  const int hi = lane >> 4;
#pragma unroll
  for (int r = 0; r < 8; ++r) {
    const int h = h0 + (hi << 3) + r;
    float v = acc[r];
    if (h == g) v -= NEGC;
    eH[(((long)b * H_ + h) * W_ + w) * H_ + g] = v;
  }
}

// ---------------------------------------------------------------------------
// Kernel 2b: energy_W[b,h,w,v'] = sum_c q[c,h,w] k[c,h,v']   (per (b,h))
// Q from Qt2 (b,h,w,c); K from Kt2 (b,h,v',c): contiguous in c.
// ---------------------------------------------------------------------------
__global__ __launch_bounds__(256) void energyW_kernel(
    const __bf16* __restrict__ Qt2, const __bf16* __restrict__ Kt2,
    float* __restrict__ eW) {
  const int lane = threadIdx.x & 31;
  const int wid  = blockIdx.x * 8 + (threadIdx.x >> 5);
  const int tile = wid % 36;
  const int h    = (wid / 36) % H_;
  const int b    = wid / (36 * H_);
  const int w0   = (tile / 6) * 16;
  const int v0   = (tile % 6) * 16;

  const __bf16* Qp = Qt2 + ((long)(b * H_ + h) * W_ + w0) * CQ_;   // A(m=w,k=c)
  const __bf16* Kp = Kt2 + ((long)(b * H_ + h) * W_ + v0) * CQ_;   // B(k=c,n=v')

  v8f acc = {};
#pragma unroll
  for (int k0 = 0; k0 < CQ_; k0 += 32) {
    v16bf a  = ldA_bfc(Qp + k0, (long)CQ_, lane);
    v16bf bm = ldB_bfc(Kp + k0, (long)CQ_, lane);
    acc = wmma_bf16(a, bm, acc);
  }

  const int vv = v0 + (lane & 15);
  const int hi = lane >> 4;
#pragma unroll
  for (int r = 0; r < 8; ++r) {
    const int w = w0 + (hi << 3) + r;
    eW[(((long)b * H_ + h) * W_ + w) * W_ + vv] = acc[r];
  }
}

// ---------------------------------------------------------------------------
// Kernel 3: softmax over concat[eH row (96), eW row (96)] per pixel, in place
// (fp32).  One thread per pixel; rows contiguous and disjoint per thread.
// ---------------------------------------------------------------------------
__global__ __launch_bounds__(256) void softmax_kernel(float* __restrict__ eH,
                                                      float* __restrict__ eW) {
  const long p  = (long)blockIdx.x * 256 + threadIdx.x;  // B*H*W = 73728 exactly
  float* rH = eH + p * H_;
  float* rW = eW + p * W_;

  float mx = -3.4e38f;
  for (int i = 0; i < H_; ++i) mx = fmaxf(mx, rH[i]);
  for (int i = 0; i < W_; ++i) mx = fmaxf(mx, rW[i]);

  float s = 0.f;
  for (int i = 0; i < H_; ++i) s += __expf(rH[i] - mx);
  for (int i = 0; i < W_; ++i) s += __expf(rW[i] - mx);
  const float inv = 1.f / s;

  for (int i = 0; i < H_; ++i) rH[i] = __expf(rH[i] - mx) * inv;
  for (int i = 0; i < W_; ++i) rW[i] = __expf(rW[i] - mx) * inv;
}

// ---------------------------------------------------------------------------
// Kernel 4: out_H[b,c,h,w] = sum_g v[c,g,w] attH[h,w,g]   (per (b,w))
// V from Vt (b,w,c,g): contiguous in g.  attH (b,h,w,g): contiguous in g.
// Raw f32 sums staged in d_out; finalized by outW_kernel.
// ---------------------------------------------------------------------------
__global__ __launch_bounds__(256) void outH_kernel(
    const __bf16* __restrict__ Vt, const float* __restrict__ attH,
    float* __restrict__ outb) {
  const int lane = threadIdx.x & 31;
  const int wid  = blockIdx.x * 8 + (threadIdx.x >> 5);
  const int tile = wid % 192;                     // 32 c-tiles x 6 h-tiles
  const int w    = (wid / 192) % W_;
  const int b    = wid / (192 * W_);
  const int c0   = (tile / 6) * 16;
  const int h0   = (tile % 6) * 16;

  const __bf16* Vp = Vt + ((long)(b * W_ + w) * C_ + c0) * H_;         // A(m=c,k=g)
  const float*  Ap = attH + (((long)b * H_ + h0) * W_ + w) * (long)H_; // B(k=g,n=h)

  v8f acc = {};
#pragma unroll
  for (int k0 = 0; k0 < H_; k0 += 32) {
    v16bf a  = ldA_bfc(Vp + k0, (long)H_, lane);
    v16bf bm = ldB_fc(Ap + k0, (long)W_ * H_, lane);
    acc = wmma_bf16(a, bm, acc);
  }

  const int h  = h0 + (lane & 15);
  const int hi = lane >> 4;
#pragma unroll
  for (int r = 0; r < 8; ++r) {
    const int c = c0 + (hi << 3) + r;
    outb[((long)b * C_ + c) * HW_ + (long)h * W_ + w] = acc[r];
  }
}

// ---------------------------------------------------------------------------
// Kernel 5: out_W + finalize:  out = gamma*(outH + outW) + x   (per (b,h))
// V natural (b,c,h,v'): contiguous in v'.  attW (b,h,w,v'): contiguous in v'.
// ---------------------------------------------------------------------------
__global__ __launch_bounds__(256) void outW_kernel(
    const __bf16* __restrict__ Vn, const float* __restrict__ attW,
    const float* __restrict__ x, const float* __restrict__ gamma,
    float* __restrict__ outb) {
  const int lane = threadIdx.x & 31;
  const int wid  = blockIdx.x * 8 + (threadIdx.x >> 5);
  const int tile = wid % 192;                     // 32 c-tiles x 6 w-tiles
  const int h    = (wid / 192) % H_;
  const int b    = wid / (192 * H_);
  const int c0   = (tile / 6) * 16;
  const int w0   = (tile % 6) * 16;

  const __bf16* Vp = Vn + ((long)b * C_ + c0) * HW_ + (long)h * W_;    // A(m=c,k=v')
  const float*  Ap = attW + (((long)b * H_ + h) * W_ + w0) * (long)W_; // B(k=v',n=w)

  v8f acc = {};
#pragma unroll
  for (int k0 = 0; k0 < W_; k0 += 32) {
    v16bf a  = ldA_bfc(Vp + k0, (long)HW_, lane);
    v16bf bm = ldB_fc(Ap + k0, (long)W_, lane);
    acc = wmma_bf16(a, bm, acc);
  }

  const float gm = gamma[0];
  const int   w  = w0 + (lane & 15);
  const int   hi = lane >> 4;
#pragma unroll
  for (int r = 0; r < 8; ++r) {
    const int  c   = c0 + (hi << 3) + r;
    const long idx = ((long)b * C_ + c) * HW_ + (long)h * W_ + w;
    outb[idx] = gm * (outb[idx] + acc[r]) + x[idx];
  }
}

// ---------------------------------------------------------------------------
extern "C" void kernel_launch(void* const* d_in, const int* in_sizes, int n_in,
                              void* d_out, int out_size, void* d_ws,
                              size_t ws_size, hipStream_t stream) {
  const float* x     = (const float*)d_in[0];
  const float* Wq    = (const float*)d_in[1];
  const float* bq    = (const float*)d_in[2];
  const float* Wk    = (const float*)d_in[3];
  const float* bk    = (const float*)d_in[4];
  const float* Wv    = (const float*)d_in[5];
  const float* bv    = (const float*)d_in[6];
  const float* gamma = (const float*)d_in[7];
  float* outb = (float*)d_out;

  // workspace layout (~322 MB), all offsets 64B-aligned
  char*   ws  = (char*)d_ws;
  size_t  o   = 0;
  __bf16* Xt  = (__bf16*)(ws + o); o += (size_t)B_ * HW_ * C_ * 2;   // 75.5 MB
  __bf16* Qt1 = (__bf16*)(ws + o); o += (size_t)B_ * HW_ * CQ_ * 2;  //  9.4 MB
  __bf16* Qt2 = (__bf16*)(ws + o); o += (size_t)B_ * HW_ * CQ_ * 2;
  __bf16* Kt1 = (__bf16*)(ws + o); o += (size_t)B_ * HW_ * CQ_ * 2;
  __bf16* Kt2 = (__bf16*)(ws + o); o += (size_t)B_ * HW_ * CQ_ * 2;
  __bf16* Vn  = (__bf16*)(ws + o); o += (size_t)B_ * C_ * HW_ * 2;   // 75.5 MB
  __bf16* Vt  = (__bf16*)(ws + o); o += (size_t)B_ * C_ * HW_ * 2;
  float*  eH  = (float*) (ws + o); o += (size_t)B_ * H_ * W_ * H_ * 4;
  float*  eW  = (float*) (ws + o); o += (size_t)B_ * H_ * W_ * W_ * 4;
  __bf16* Wqb = (__bf16*)(ws + o); o += (size_t)CQ_ * C_ * 2;        // 64KB
  __bf16* Wkb = (__bf16*)(ws + o); o += (size_t)CQ_ * C_ * 2;
  __bf16* Wvb = (__bf16*)(ws + o); o += (size_t)C_ * C_ * 2;         // 512KB

  // weights -> bf16 (tiny, once per call)
  cvt_kernel<<<(CQ_ * C_ + 255) / 256, 256, 0, stream>>>(Wq, Wqb, CQ_ * C_);
  cvt_kernel<<<(CQ_ * C_ + 255) / 256, 256, 0, stream>>>(Wk, Wkb, CQ_ * C_);
  cvt_kernel<<<(C_ * C_ + 255) / 256, 256, 0, stream>>>(Wv, Wvb, C_ * C_);

  transpose_kernel<<<36864, 256, 0, stream>>>(x, Xt);   // 8 * 16 * 288 tiles
  proj_kernel   <<<23040, 256, 0, stream>>>(Xt, Wqb, bq, Wkb, bk, Wvb, bv,
                                            Qt1, Qt2, Kt1, Kt2, Vn, Vt);
  energyH_kernel<<< 3456, 256, 0, stream>>>(Qt1, Kt1, eH);
  energyW_kernel<<< 3456, 256, 0, stream>>>(Qt2, Kt2, eW);
  softmax_kernel<<<  288, 256, 0, stream>>>(eH, eW);    // 73728 pixels exactly
  outH_kernel   <<<18432, 256, 0, stream>>>(Vt, eH, outb);
  outW_kernel   <<<18432, 256, 0, stream>>>(Vn, eW, x, gamma, outb);
}
